// Block_46669114638369
// MI455X (gfx1250) — matrix operand order repaired and checked by reading, not compiled, not run on hardware
//
#include <hip/hip_runtime.h>
#include <cstddef>
#include <cstdint>

// ---------------------------------------------------------------------------
// Problem constants (match the JAX reference)
// ---------------------------------------------------------------------------
#define D_MODEL 1024
#define D_INNER 2048
#define D_STATE 16
#define BATCH   2
#define SEQ     2048
#define ML      (BATCH * SEQ)          // 4096 flattened rows

typedef __attribute__((ext_vector_type(16))) __bf16         v16bf;
typedef __attribute__((ext_vector_type(8)))  float          v8f;
typedef __attribute__((ext_vector_type(8)))  unsigned short ushort8;
typedef __attribute__((ext_vector_type(4)))  unsigned int   v4u;
typedef __attribute__((ext_vector_type(8)))  int            v8i;
typedef __attribute__((ext_vector_type(4)))  int            v4i;

union Frag16 {
    v16bf          v;
    ushort8        g[2];
    unsigned short u[16];
};

// LDS row stride (ushorts): 32 payload + 8 pad = 80B rows, 16B aligned groups.
// TDM reproduces this via pad_enable (pad 4 DWORDs after every 16 DWORDs).
constexpr int LDS_RS = 40;

// ---------------------------------------------------------------------------
// Helpers
// ---------------------------------------------------------------------------
__device__ __forceinline__ v8f wmma_bf16(const Frag16& a, const Frag16& b, v8f c) {
    return __builtin_amdgcn_wmma_f32_16x16x32_bf16(
        false, a.v, false, b.v, (short)0, c, false, false);
}

// fp32 -> (bf16 hi, bf16 lo residual); hi+lo carries ~16 mantissa bits
__device__ __forceinline__ void split_bf16(float f, unsigned short& hi, unsigned short& lo) {
    unsigned int bits = __float_as_uint(f);
    hi = (unsigned short)(bits >> 16);
    float fh = __uint_as_float(bits & 0xFFFF0000u);
    lo = (unsigned short)(__float_as_uint(f - fh) >> 16);
}

// A fragment (16x32, MxK). lanes 0-15: row M=lane, K 0..7 & 16..23;
// lanes 16-31: row M=lane-16, K 8..15 & 24..31.  LDS is [row][k].
__device__ __forceinline__ void load_frag_a(const unsigned short* lds, int rbase, Frag16& f) {
    const int lane = threadIdx.x & 31;
    const int hi   = lane >> 4;
    const int m    = lane & 15;
    const unsigned short* p = lds + (rbase + m) * LDS_RS;
    f.g[0] = *(const ushort8*)(p + 8 * hi);
    f.g[1] = *(const ushort8*)(p + 16 + 8 * hi);
}

// B fragment (32x16, KxN). lanes 0-15 K 0..7 & 16..23, lanes 16-31 K 8..15 & 24..31, N=lane&15.
// LDS holds B transposed: [col][k], 8 contiguous K values per group.
__device__ __forceinline__ void load_frag_b(const unsigned short* lds, int cbase, Frag16& f) {
    const int lane = threadIdx.x & 31;
    const int hi   = lane >> 4;
    const int n    = lane & 15;
    const unsigned short* p = lds + (cbase + n) * LDS_RS;
    f.g[0] = *(const ushort8*)(p + 8 * hi);
    f.g[1] = *(const ushort8*)(p + 16 + 8 * hi);
}

// Epilogue activations: 0 = none, 1 = SELU, 2 = softplus
template <int ACT>
__device__ __forceinline__ float act(float x) {
    if constexpr (ACT == 1) {
        const float kScale = 1.0507009873554805f;
        const float kAlpha = 1.6732632423543772f;
        return x > 0.f ? kScale * x : kScale * kAlpha * (__expf(x) - 1.f);
    } else if constexpr (ACT == 2) {
        return (x > 20.f) ? x : log1pf(__expf(x));
    } else {
        return x;
    }
}

// ---------------------------------------------------------------------------
// Tensor Data Mover: async 2D tile load (global -> LDS), data_size = 2 bytes.
// Descriptor packing per CDNA5 ISA.
// pad_enable: +4 DWORDs (16B) after every 16 DWORDs (64B) => LDS row stride 80B
// which is exactly LDS_RS(40) ushorts for a 32-element tile row.
// ---------------------------------------------------------------------------
__device__ __forceinline__ void tdm_load_2d(
    unsigned int lds_addr, const void* gaddr,
    unsigned int tensor_d0, unsigned int tensor_d1, unsigned long long stride_e,
    unsigned int tile_d0, unsigned int tile_d1) {
    unsigned long long ga = (unsigned long long)(uintptr_t)gaddr;

    v4u g0;
    g0[0] = 1u;                                     // count=1, user descriptor
    g0[1] = lds_addr;                               // LDS byte address
    g0[2] = (unsigned int)ga;                       // global addr [31:0]
    g0[3] = (unsigned int)((ga >> 32) & 0x1FFFFFFu) // global addr [56:32]
            | 0x80000000u;                          // type = 2 ("image")

    v8i g1;
    g1[0] = (int)((1u << 16)    // data_size = 2 bytes
                | (1u << 20)    // pad_enable
                | (3u << 22)    // pad_interval: every 16 DWORDs
                | (3u << 25));  // pad_amount: 4 DWORDs
    g1[1] = (int)((tensor_d0 & 0xFFFFu) << 16);
    g1[2] = (int)(((tensor_d0 >> 16) & 0xFFFFu) | ((tensor_d1 & 0xFFFFu) << 16));
    g1[3] = (int)(((tensor_d1 >> 16) & 0xFFFFu) | ((tile_d0 & 0xFFFFu) << 16));
    g1[4] = (int)(tile_d1 & 0xFFFFu);               // tile_dim1 (tile_dim2 = 0)
    g1[5] = (int)(unsigned int)(stride_e & 0xFFFFFFFFu);
    g1[6] = (int)(unsigned int)((stride_e >> 32) & 0xFFFFu);  // dim1_stride = 0
    g1[7] = 0;

    v4i z = {0, 0, 0, 0};
#if defined(__clang_major__) && (__clang_major__ >= 23)
    v8i z8 = {0, 0, 0, 0, 0, 0, 0, 0};
    __builtin_amdgcn_tensor_load_to_lds(g0, g1, z, z, z8, 0);
#else
    __builtin_amdgcn_tensor_load_to_lds(g0, g1, z, z, 0);
#endif
}

__device__ __forceinline__ unsigned int lds_addr_of(const void* p) {
    // generic shared pointer: low 32 bits are the LDS byte address
    return (unsigned int)(uintptr_t)p;
}

// ---------------------------------------------------------------------------
// Pre-pass: elementwise fp32 -> bf16 hi/lo planes (same layout)
// ---------------------------------------------------------------------------
__global__ __launch_bounds__(256) void split_plane(
    const float* __restrict__ X, unsigned short* __restrict__ Xh,
    unsigned short* __restrict__ Xl, size_t n) {
    size_t i = ((size_t)blockIdx.x * 256 + threadIdx.x) * 4;
    if (i + 3 >= n) return;
    float4 v = *(const float4*)(X + i);
    const float f[4] = {v.x, v.y, v.z, v.w};
#pragma unroll
    for (int j = 0; j < 4; ++j) {
        unsigned short h, l;
        split_bf16(f[j], h, l);
        Xh[i + j] = h;
        Xl[i + j] = l;
    }
}

// ---------------------------------------------------------------------------
// Pre-pass: fp32 W[R][C] -> transposed bf16 hi/lo planes T[C][R]
// ---------------------------------------------------------------------------
__global__ __launch_bounds__(256) void split_transpose(
    const float* __restrict__ W, unsigned short* __restrict__ Th,
    unsigned short* __restrict__ Tl, int R, int C) {
    __shared__ float tile[32][33];
    const int c0 = blockIdx.x * 32;
    const int r0 = blockIdx.y * 32;
    const int tx = threadIdx.x & 31;
    const int ty = threadIdx.x >> 5;   // 0..7
#pragma unroll
    for (int i = 0; i < 32; i += 8) {
        const int r = r0 + ty + i, c = c0 + tx;
        tile[ty + i][tx] = (r < R && c < C) ? W[(size_t)r * C + c] : 0.f;
    }
    __syncthreads();
#pragma unroll
    for (int i = 0; i < 32; i += 8) {
        const int tr = c0 + ty + i;    // out row = original col
        const int tc = r0 + tx;        // out col = original row
        if (tr < C && tc < R) {
            unsigned short h, l;
            split_bf16(tile[tx][ty + i], h, l);
            Th[(size_t)tr * R + tc] = h;
            Tl[(size_t)tr * R + tc] = l;
        }
    }
}

// ---------------------------------------------------------------------------
// TDM-fed bf16x3 split GEMM:  C[M,N] = act(A[M,K] @ B[K,N] + bias[N])
// Operands are preconverted bf16 hi/lo planes; B planes are TRANSPOSED [N][K].
// 256 threads = 8 waves; 128x128 tile; BK=32; wave -> 32x64 slab (24 WMMA/step).
// Wave 0 drives 4 TENSOR_LOAD_TO_LDS per k-step into the idle LDS buffer;
// s_wait_tensorcnt(4) overlaps current-buffer WMMAs with next-buffer DMA.
// Optionally also emits split planes of the output (for chained GEMMs).
// ---------------------------------------------------------------------------
template <int ACT, bool WRITE_SPLIT>
__global__ __launch_bounds__(256) void gemm_tdm(
    const unsigned short* __restrict__ Ahp, const unsigned short* __restrict__ Alp,
    const unsigned short* __restrict__ Bhp, const unsigned short* __restrict__ Blp,
    const float* __restrict__ bias, float* __restrict__ C,
    unsigned short* __restrict__ Chp, unsigned short* __restrict__ Clp,
    int M, int N, int K) {
    constexpr int BM = 128, BN = 128, BK = 32;
    __shared__ __align__(16) unsigned short Ah[2][BM * LDS_RS];
    __shared__ __align__(16) unsigned short Al[2][BM * LDS_RS];
    __shared__ __align__(16) unsigned short Bh[2][BN * LDS_RS];
    __shared__ __align__(16) unsigned short Bl[2][BN * LDS_RS];

    const int tid   = threadIdx.x;
    const int wave  = tid >> 5;
    const int wm    = wave & 3;   // 0..3 : 32-row slab
    const int wn    = wave >> 2;  // 0..1 : 64-col slab
    const int tileM = blockIdx.y * BM;
    const int tileN = blockIdx.x * BN;

    v8f acc[2][4];
#pragma unroll
    for (int mt = 0; mt < 2; ++mt)
#pragma unroll
        for (int nt = 0; nt < 4; ++nt) acc[mt][nt] = (v8f){};

    // prologue: DMA k-tile 0 into buffer 0 (wave 0 only; TENSORcnt tracks it)
    if (tid < 32) {
        tdm_load_2d(lds_addr_of(&Ah[0][0]), Ahp + (size_t)tileM * K, K, M, K, BK, BM);
        tdm_load_2d(lds_addr_of(&Al[0][0]), Alp + (size_t)tileM * K, K, M, K, BK, BM);
        tdm_load_2d(lds_addr_of(&Bh[0][0]), Bhp + (size_t)tileN * K, K, N, K, BK, BN);
        tdm_load_2d(lds_addr_of(&Bl[0][0]), Blp + (size_t)tileN * K, K, N, K, BK, BN);
    }

    int p = 0;
    for (int k0 = 0; k0 < K; k0 += BK) {
        if (tid < 32) {
            if (k0 + BK < K) {
                const int q = p ^ 1, k1 = k0 + BK;
                tdm_load_2d(lds_addr_of(&Ah[q][0]), Ahp + (size_t)tileM * K + k1, K, M, K, BK, BM);
                tdm_load_2d(lds_addr_of(&Al[q][0]), Alp + (size_t)tileM * K + k1, K, M, K, BK, BM);
                tdm_load_2d(lds_addr_of(&Bh[q][0]), Bhp + (size_t)tileN * K + k1, K, N, K, BK, BN);
                tdm_load_2d(lds_addr_of(&Bl[q][0]), Blp + (size_t)tileN * K + k1, K, N, K, BK, BN);
                __builtin_amdgcn_s_wait_tensorcnt(4);  // buffer p's 4 loads done
            } else {
                __builtin_amdgcn_s_wait_tensorcnt(0);
            }
        }
        __syncthreads();  // LDS buffer p ready for all waves

        Frag16 fa[2][2];  // [m-subtile][hi/lo]
        load_frag_a(Ah[p], wm * 32,      fa[0][0]);
        load_frag_a(Al[p], wm * 32,      fa[0][1]);
        load_frag_a(Ah[p], wm * 32 + 16, fa[1][0]);
        load_frag_a(Al[p], wm * 32 + 16, fa[1][1]);
#pragma unroll
        for (int nt = 0; nt < 4; ++nt) {
            Frag16 fbh, fbl;
            load_frag_b(Bh[p], wn * 64 + nt * 16, fbh);
            load_frag_b(Bl[p], wn * 64 + nt * 16, fbl);
#pragma unroll
            for (int mt = 0; mt < 2; ++mt) {
                // (Ah+Al)(Bh+Bl) ~= Ah*Bh + Ah*Bl + Al*Bh
                acc[mt][nt] = wmma_bf16(fa[mt][0], fbh, acc[mt][nt]);
                acc[mt][nt] = wmma_bf16(fa[mt][0], fbl, acc[mt][nt]);
                acc[mt][nt] = wmma_bf16(fa[mt][1], fbh, acc[mt][nt]);
            }
        }
        __syncthreads();  // all reads of buffer p done before TDM reuses it
        p ^= 1;
    }

    // C/D layout: VGPR r, lane l -> row = r + 8*(l>>4), col = l&15
    const int lane = tid & 31;
    const int hi   = lane >> 4;
    const int cn   = lane & 15;
#pragma unroll
    for (int mt = 0; mt < 2; ++mt) {
        const int gm0 = tileM + wm * 32 + mt * 16 + hi * 8;
#pragma unroll
        for (int nt = 0; nt < 4; ++nt) {
            const int gn = tileN + wn * 64 + nt * 16 + cn;
            const float bs = bias[gn];
#pragma unroll
            for (int r = 0; r < 8; ++r) {
                const float v = act<ACT>(acc[mt][nt][r] + bs);
                const size_t idx = (size_t)(gm0 + r) * N + gn;
                C[idx] = v;
                if constexpr (WRITE_SPLIT) {
                    unsigned short h, l;
                    split_bf16(v, h, l);
                    Chp[idx] = h;
                    Clp[idx] = l;
                }
            }
        }
    }
}

// ---------------------------------------------------------------------------
// Fused B/C projection:  Bm = H@W_B + b_B, Cm = H@W_C + b_C  (N = 16 each)
// ---------------------------------------------------------------------------
__global__ __launch_bounds__(256) void gemm_bc(
    const float* __restrict__ H,
    const float* __restrict__ WB, const float* __restrict__ bB,
    const float* __restrict__ WC, const float* __restrict__ bC,
    float* __restrict__ Bm, float* __restrict__ Cm, int M, int K) {
    constexpr int BM = 128, BK = 32;
    __shared__ __align__(16) unsigned short Hh[BM * LDS_RS];
    __shared__ __align__(16) unsigned short Hl[BM * LDS_RS];
    __shared__ __align__(16) unsigned short Wh[2][16 * LDS_RS];
    __shared__ __align__(16) unsigned short Wl[2][16 * LDS_RS];

    const int tid   = threadIdx.x;
    const int wave  = tid >> 5;
    const int tileM = blockIdx.x * BM;

    const int hr = tid >> 1;
    const int hc = (tid & 1) * 16;
    const float* gH = H + (size_t)(tileM + hr) * K + hc;

    const int wsel = tid >> 7;
    const int t2   = tid & 127;
    const int kr   = (t2 * 4) >> 4;
    const int nc   = (t2 * 4) & 15;
    const float* gW = (wsel ? WC : WB) + (size_t)kr * D_STATE + nc;

    v8f accB = {};
    v8f accC = {};

    for (int k0 = 0; k0 < K; k0 += BK) {
        float4 h0 = *(const float4*)(gH);
        float4 h1 = *(const float4*)(gH + 4);
        float4 h2 = *(const float4*)(gH + 8);
        float4 h3 = *(const float4*)(gH + 12);
        float4 wv = *(const float4*)(gW);
        if (k0 + BK < K) {
            __builtin_prefetch(gH + BK, 0, 1);
            __builtin_prefetch(gW + (size_t)BK * D_STATE, 0, 1);
        }
        const float hv[16] = {h0.x, h0.y, h0.z, h0.w, h1.x, h1.y, h1.z, h1.w,
                              h2.x, h2.y, h2.z, h2.w, h3.x, h3.y, h3.z, h3.w};
#pragma unroll
        for (int i = 0; i < 16; ++i) {
            unsigned short hh, hl;
            split_bf16(hv[i], hh, hl);
            Hh[hr * LDS_RS + hc + i] = hh;
            Hl[hr * LDS_RS + hc + i] = hl;
        }
        const float wvv[4] = {wv.x, wv.y, wv.z, wv.w};
#pragma unroll
        for (int i = 0; i < 4; ++i) {
            unsigned short wh, wl;
            split_bf16(wvv[i], wh, wl);
            Wh[wsel][(nc + i) * LDS_RS + kr] = wh;
            Wl[wsel][(nc + i) * LDS_RS + kr] = wl;
        }
        __syncthreads();

        Frag16 fh, fl, fbh, fbl, fch, fcl;
        load_frag_a(Hh, wave * 16, fh);
        load_frag_a(Hl, wave * 16, fl);
        load_frag_b(Wh[0], 0, fbh);
        load_frag_b(Wl[0], 0, fbl);
        load_frag_b(Wh[1], 0, fch);
        load_frag_b(Wl[1], 0, fcl);

        accB = wmma_bf16(fh, fbh, accB);
        accB = wmma_bf16(fh, fbl, accB);
        accB = wmma_bf16(fl, fbh, accB);
        accC = wmma_bf16(fh, fch, accC);
        accC = wmma_bf16(fh, fcl, accC);
        accC = wmma_bf16(fl, fch, accC);

        __syncthreads();
        gH += BK;
        gW += (size_t)BK * D_STATE;
    }

    const int lane = tid & 31;
    const int hi   = lane >> 4;
    const int n    = lane & 15;
    const int gm0  = tileM + wave * 16 + hi * 8;
    const float bBn = bB[n];
    const float bCn = bC[n];
#pragma unroll
    for (int r = 0; r < 8; ++r) {
        const size_t row = (size_t)(gm0 + r) * D_STATE;
        Bm[row + n] = accB[r] + bBn;
        Cm[row + n] = accC[r] + bCn;
    }
}

// ---------------------------------------------------------------------------
// Selective-scan: one thread per (b, d) channel, 16 states in registers.
// Emits y directly as bf16 hi/lo planes (A-operand of the output GEMM).
// ---------------------------------------------------------------------------
__global__ __launch_bounds__(256) void scan_kernel(
    const float* __restrict__ delta, const float* __restrict__ H,
    const float* __restrict__ Bm, const float* __restrict__ Cm,
    const float* __restrict__ A_log, const float* __restrict__ Dp,
    unsigned short* __restrict__ Yh, unsigned short* __restrict__ Yl) {
    __shared__ float bc[2 * D_STATE];

    const int blocksPerBatch = D_INNER / 256;  // 8
    const int b = blockIdx.x / blocksPerBatch;
    const int d = (blockIdx.x % blocksPerBatch) * 256 + threadIdx.x;

    float a[D_STATE], s[D_STATE];
#pragma unroll
    for (int n = 0; n < D_STATE; ++n) {
        a[n] = -__expf(A_log[(size_t)d * D_STATE + n]);
        s[n] = 0.f;
    }
    const float dparam = Dp[d];
    const size_t base = (size_t)b * SEQ;

    for (int t = 0; t < SEQ; ++t) {
        if (threadIdx.x < 2 * D_STATE) {
            const float* src = (threadIdx.x < D_STATE)
                ? (Bm + (base + t) * D_STATE + threadIdx.x)
                : (Cm + (base + t) * D_STATE + (threadIdx.x - D_STATE));
            bc[threadIdx.x] = *src;
        }
        __syncthreads();

        const float dt = delta[(base + t) * D_INNER + d];
        const float hv = H[(base + t) * D_INNER + d];
        float y = 0.f;
#pragma unroll
        for (int n = 0; n < D_STATE; ++n) {
            const float dA = __expf(dt * a[n]);
            s[n] = dA * s[n] + (dt * bc[n]) * hv;
            y += s[n] * bc[D_STATE + n];
        }
        unsigned short yh, yl;
        split_bf16(y + hv * dparam, yh, yl);
        const size_t idx = (base + t) * D_INNER + d;
        Yh[idx] = yh;
        Yl[idx] = yl;
        __syncthreads();
    }
}

// ---------------------------------------------------------------------------
// Launcher
// ---------------------------------------------------------------------------
extern "C" void kernel_launch(void* const* d_in, const int* in_sizes, int n_in,
                              void* d_out, int out_size, void* d_ws, size_t ws_size,
                              hipStream_t stream) {
    (void)in_sizes; (void)n_in; (void)out_size; (void)ws_size;

    const float* x       = (const float*)d_in[0];
    const float* W_in    = (const float*)d_in[1];
    const float* b_in    = (const float*)d_in[2];
    const float* A_log   = (const float*)d_in[3];
    const float* W_B     = (const float*)d_in[4];
    const float* b_B     = (const float*)d_in[5];
    const float* W_C     = (const float*)d_in[6];
    const float* b_C     = (const float*)d_in[7];
    const float* W_delta = (const float*)d_in[8];
    const float* b_delta = (const float*)d_in[9];
    const float* D_param = (const float*)d_in[10];
    const float* W_out   = (const float*)d_in[11];
    const float* b_out   = (const float*)d_in[12];
    float* out = (float*)d_out;

    // workspace carve-up
    char* ws = (char*)d_ws;
    const size_t HB  = (size_t)ML * D_INNER * sizeof(float);          // 32 MB
    const size_t SB  = (size_t)ML * D_STATE * sizeof(float);          // 256 KB
    const size_t XS  = (size_t)ML * D_MODEL * 2;                      // 8 MB
    const size_t HS  = (size_t)ML * D_INNER * 2;                      // 16 MB
    const size_t WIS = (size_t)D_MODEL * D_INNER * 2;                 // 4 MB
    const size_t WDS = (size_t)D_INNER * D_INNER * 2;                 // 8 MB
    const size_t WOS = (size_t)D_INNER * D_MODEL * 2;                 // 4 MB

    size_t off = 0;
    float* h     = (float*)(ws + off); off += HB;
    float* delta = (float*)(ws + off); off += HB;
    float* Bm    = (float*)(ws + off); off += SB;
    float* Cm    = (float*)(ws + off); off += SB;
    unsigned short* xh  = (unsigned short*)(ws + off); off += XS;
    unsigned short* xl  = (unsigned short*)(ws + off); off += XS;
    unsigned short* hh  = (unsigned short*)(ws + off); off += HS;
    unsigned short* hl  = (unsigned short*)(ws + off); off += HS;
    unsigned short* yh  = (unsigned short*)(ws + off); off += HS;
    unsigned short* yl  = (unsigned short*)(ws + off); off += HS;
    unsigned short* wih = (unsigned short*)(ws + off); off += WIS;  // W_in^T  [2048][1024]
    unsigned short* wil = (unsigned short*)(ws + off); off += WIS;
    unsigned short* wdh = (unsigned short*)(ws + off); off += WDS;  // W_delta^T [2048][2048]
    unsigned short* wdl = (unsigned short*)(ws + off); off += WDS;
    unsigned short* woh = (unsigned short*)(ws + off); off += WOS;  // W_out^T [1024][2048]
    unsigned short* wol = (unsigned short*)(ws + off); off += WOS;

    const dim3 blk(256);

    // 0) one-time operand conversion to bf16 hi/lo planes
    split_plane<<<dim3((unsigned)((size_t)ML * D_MODEL / 1024)), blk, 0, stream>>>(
        x, xh, xl, (size_t)ML * D_MODEL);
    split_transpose<<<dim3(D_INNER / 32, D_MODEL / 32), blk, 0, stream>>>(
        W_in, wih, wil, D_MODEL, D_INNER);
    split_transpose<<<dim3(D_INNER / 32, D_INNER / 32), blk, 0, stream>>>(
        W_delta, wdh, wdl, D_INNER, D_INNER);
    split_transpose<<<dim3(D_MODEL / 32, D_INNER / 32), blk, 0, stream>>>(
        W_out, woh, wol, D_INNER, D_MODEL);

    // 1) h = selu(x @ W_in + b_in)  (also emits h split planes)
    gemm_tdm<1, true><<<dim3(D_INNER / 128, ML / 128), blk, 0, stream>>>(
        xh, xl, wih, wil, b_in, h, hh, hl, ML, D_INNER, D_MODEL);

    // 2) delta = softplus(h @ W_delta + b_delta)
    gemm_tdm<2, false><<<dim3(D_INNER / 128, ML / 128), blk, 0, stream>>>(
        hh, hl, wdh, wdl, b_delta, delta, nullptr, nullptr, ML, D_INNER, D_INNER);

    // 3) Bm / Cm projections
    gemm_bc<<<dim3(ML / 128), blk, 0, stream>>>(
        h, W_B, b_B, W_C, b_C, Bm, Cm, ML, D_INNER);

    // 4) selective scan -> y split planes
    scan_kernel<<<dim3(BATCH * (D_INNER / 256)), blk, 0, stream>>>(
        delta, h, Bm, Cm, A_log, D_param, yh, yl);

    // 5) out = y @ W_out + b_out
    gemm_tdm<0, false><<<dim3(D_MODEL / 128, ML / 128), blk, 0, stream>>>(
        yh, yl, woh, wol, b_out, out, nullptr, nullptr, ML, D_MODEL, D_INNER);
}
